// ConnectedComponetsHead_69286412419667
// MI455X (gfx1250) — matrix operand order repaired and checked by reading, not compiled, not run on hardware
//
#include <hip/hip_runtime.h>
#include <hip/hip_bf16.h>
#include <math.h>

// ---------------- problem constants ----------------
#define BB 2
#define CC 64
#define DD 8
#define HH 128
#define WW 128
#define VV (DD * HH * WW)          // 131072
#define LL 101
#define NCONV 4
#define KD 64
#define NLM 20
#define NN 120                      // 100 segment means + 20 learned
#define EPSF 1e-3f

typedef __attribute__((ext_vector_type(16))) __bf16 v16bf;
typedef __attribute__((ext_vector_type(8)))  float  v8f;

// Fragment as 8 dwords (two bf16 per dword, element 2j = low half).
union Frag16 {
  v16bf v;
  unsigned int u[8];
};

__device__ __forceinline__ unsigned int pack_bf16x2(float lo, float hi) {
  union { __bf16 h[2]; unsigned int u; } p;
  p.h[0] = (__bf16)lo;
  p.h[1] = (__bf16)hi;
  return p.u;
}

// -------- WMMA bf16 fragment layouts (CDNA5 wave32) --------
// A (16x32, 16-bit): lane L row m=L&15; dword j holds K0,K0+1 with
//   K0 = (j>=4?16:0) + 8*(L>=16) + 2*(j&3)
// B (32x16, 16-bit): lane L col n=L&15; dword j holds K0,K0+1 with K0 as for A
// C/D (16x16 f32):   lane L col n=L&15; reg r -> m = r + 8*(L>=16)

__device__ __forceinline__ int a_pair_k0(int j, int half) {
  return ((j & 4) ? 16 : 0) + half * 8 + ((j & 3) << 1);
}

// =====================================================================
// Kernel 0: pack conv weights fp32 [layer][oc][ic][1][3][3] ->
//   uint32 pairs [layer][tap][icpair][oc]  (one dword = ic 2p, 2p+1)
// =====================================================================
__global__ __launch_bounds__(256) void pack_weights_kernel(
    const float* __restrict__ cw, unsigned int* __restrict__ wpack) {
  int idx = blockIdx.x * 256 + threadIdx.x;
  const int total = NCONV * 9 * (CC / 2) * CC;   // 73728
  if (idx >= total) return;
  int oc = idx & 63;
  int t = idx >> 6;
  int icp = t & 31;
  t >>= 5;
  int tap = t % 9;
  int layer = t / 9;
  const float* base = cw + (((long)(layer * CC + oc) * CC) * 9) + tap;
  float lo = base[(2 * icp) * 9];
  float hi = base[(2 * icp + 1) * 9];
  wpack[idx] = pack_bf16x2(lo, hi);
}

// =====================================================================
// Kernel 1: 3x3 conv (spatial HxW, pad 1) + bias + ReLU, implicit GEMM.
// grid = B*D*H*(W/16); block = 128 (4 waves). Wave w -> out channels
// [16w,16w+16). M = oc (A = packed weights), N = 16 pixels (B = LDS patch).
// K = tap*64 + ic, 18 chunks of 32.
// LDS patch layout [row][col][ic] padded to 66 per line: ic contiguous
// (dword-pair B loads), lane stride 132B = 33 banks (conflict-free).
// =====================================================================
#define PADC 66
__global__ __launch_bounds__(128) void conv3x3_wmma_kernel(
    const float* __restrict__ in, const unsigned int* __restrict__ wpack,
    const float* __restrict__ bias, float* __restrict__ out) {
  int gid = blockIdx.x;
  int wt = gid & 7;               // W/16 = 8
  int h  = (gid >> 3) & 127;      // H
  int d  = (gid >> 10) & 7;       // D
  int b  = gid >> 13;             // B
  int w0 = wt * 16;

  __shared__ __bf16 patch[3 * 18 * PADC];
  const int tid = threadIdx.x;
  const long planeHW = (long)HH * WW;
  const float* inb = in + ((long)b * CC) * DD * planeHW + (long)d * planeHW;

  for (int idx = tid; idx < CC * 3 * 18; idx += 128) {
    int ic  = idx / 54;           // global-coalesced: col fastest
    int rem = idx % 54;
    int row = rem / 18;
    int col = rem % 18;
    int hh = h - 1 + row;
    int ww = w0 - 1 + col;
    float v = 0.0f;
    if (hh >= 0 && hh < HH && ww >= 0 && ww < WW)
      v = inb[(long)ic * DD * planeHW + hh * WW + ww];
    patch[(row * 18 + col) * PADC + ic] = (__bf16)v;
  }
  __syncthreads();

  const int wave = tid >> 5;
  const int lane = tid & 31;
  const int ocbase = wave * 16;
  const int half = (lane >> 4) & 1;
  const int l15 = lane & 15;

  v8f acc = {};
  for (int kc = 0; kc < 18; ++kc) {
    int tap = kc >> 1;
    int icbase = (kc & 1) << 5;
    int dy = tap / 3, dx = tap % 3;
    Frag16 a, bf;
    const unsigned int* wt8 = wpack + ((tap * 32 + (icbase >> 1)) * CC) + (ocbase + l15);
    const __bf16* prow = &patch[(dy * 18 + l15 + dx) * PADC + icbase];
#pragma unroll
    for (int j = 0; j < 8; ++j) {
      int icp = (a_pair_k0(j, half) >> 1);            // 0..15 within chunk
      a.u[j] = wt8[icp * CC];
      bf.u[j] = *(const unsigned int*)&prow[a_pair_k0(j, half)];   // K pair (table order)
    }
    acc = __builtin_amdgcn_wmma_f32_16x16x32_bf16(
        false, a.v, false, bf.v, (short)0, acc, false, false);
  }

  float* outb = out + ((long)b * CC) * DD * planeHW + (long)d * planeHW;
#pragma unroll
  for (int r = 0; r < 8; ++r) {
    int oc = ocbase + r + 8 * half;
    float v = acc[r] + bias[oc];
    v = v > 0.0f ? v : 0.0f;
    outb[(long)oc * DD * planeHW + h * WW + (w0 + l15)] = v;
  }
}

// =====================================================================
// Kernel 2: segment sums/counts via LDS accumulators + global atomics.
// grid = B * (V/2048); block = 256.
// =====================================================================
__global__ __launch_bounds__(256) void seg_pool_kernel(
    const float* __restrict__ x, const int* __restrict__ labels,
    float* __restrict__ sums, float* __restrict__ counts) {
  const int CHUNKS = VV / 2048;   // 64
  int b = blockIdx.x / CHUNKS;
  long v0 = (long)(blockIdx.x % CHUNKS) * 2048;

  __shared__ float ls[LL * CC];
  __shared__ float lc[LL];
  for (int i = threadIdx.x; i < LL * CC; i += 256) ls[i] = 0.0f;
  for (int i = threadIdx.x; i < LL; i += 256) lc[i] = 0.0f;
  __syncthreads();

  const float* xb = x + (long)b * CC * VV;
  const int* lb = labels + (long)b * VV;
  for (int p = threadIdx.x; p < 2048; p += 256) {
    long v = v0 + p;
    int l = lb[v];
    atomicAdd(&lc[l], 1.0f);
    for (int c = 0; c < CC; ++c)
      atomicAdd(&ls[l * CC + c], xb[(long)c * VV + v]);
  }
  __syncthreads();

  float* gs = sums + (long)b * LL * CC;
  float* gc = counts + (long)b * LL;
  for (int i = threadIdx.x; i < LL * CC; i += 256)
    if (ls[i] != 0.0f) atomicAdd(&gs[i], ls[i]);
  for (int i = threadIdx.x; i < LL; i += 256)
    if (lc[i] != 0.0f) atomicAdd(&gc[i], lc[i]);
}

// =====================================================================
// Kernel 3: inst_features [B,120,64] = segment means (labels 1..100) ++ embed
// =====================================================================
__global__ __launch_bounds__(256) void inst_features_kernel(
    const float* __restrict__ sums, const float* __restrict__ counts,
    const float* __restrict__ embed, float* __restrict__ instf) {
  int idx = blockIdx.x * 256 + threadIdx.x;
  if (idx >= BB * NN * CC) return;
  int c = idx & 63;
  int t = idx >> 6;
  int n = t % NN;
  int b = t / NN;
  float v;
  if (n < 100) {
    int l = n + 1;
    float cnt = counts[b * LL + l];
    float denom = cnt > 1.0f ? cnt : 1.0f;
    v = sums[(b * LL + l) * CC + c] / denom;
  } else {
    v = embed[(n - 100) * CC + c];
  }
  instf[idx] = v;
}

// =====================================================================
// Kernel 4: pred_kernel raw = instf @ mk_w^T + mk_b  (tiny, exact fp32)
// =====================================================================
__global__ __launch_bounds__(256) void mask_kernel_linear_kernel(
    const float* __restrict__ instf, const float* __restrict__ mkw,
    const float* __restrict__ mkb, float* __restrict__ pk) {
  int idx = blockIdx.x * 256 + threadIdx.x;
  if (idx >= BB * NN * KD) return;
  int kd = idx & 63;
  int t = idx >> 6;                 // b*120 + n
  const float* f = instf + (long)t * CC;
  const float* w = mkw + (long)kd * CC;
  float s = mkb[kd];
  for (int c = 0; c < CC; ++c) s += f[c] * w[c];
  pk[idx] = s;
}

// =====================================================================
// Kernel 5: BatchNorm1d(KD) over (B*N)=240 samples, training-mode stats.
// =====================================================================
__global__ __launch_bounds__(64) void norm_kernel_bn_kernel(
    const float* __restrict__ pk, const float* __restrict__ gamma,
    const float* __restrict__ beta, float* __restrict__ pkout) {
  int kd = threadIdx.x;
  float s = 0.0f, ss = 0.0f;
  for (int t = 0; t < BB * NN; ++t) {
    float v = pk[t * KD + kd];
    s += v;
    ss += v * v;
  }
  const float invN = 1.0f / (float)(BB * NN);
  float mu = s * invN;
  float var = ss * invN - mu * mu;
  float inv = rsqrtf(var + EPSF);
  float g = gamma[kd], bt = beta[kd];
  for (int t = 0; t < BB * NN; ++t)
    pkout[t * KD + kd] = (pk[t * KD + kd] - mu) * inv * g + bt;
}

// =====================================================================
// Kernel 6: pred_masks = pred_kernel[b] (120x64) @ mask_features[b] (64xV)
// grid = B * (V/64); block = 128 (4 waves, 16 pixels each).
// A staged in LDS as bf16 (padded to 128 rows, K contiguous -> dword reads);
// 8 M-tiles x 2 K-chunks per wave. Fused per-block sum/sum^2 partials.
// =====================================================================
__global__ __launch_bounds__(128) void bmm_wmma_kernel(
    const float* __restrict__ pk, const float* __restrict__ mf,
    float* __restrict__ pm, float* __restrict__ partial) {
  int b = blockIdx.x >> 11;                 // 2048 blocks per batch
  long v0 = (long)(blockIdx.x & 2047) * 64;

  __shared__ __bf16 apack[128 * KD];
  const int tid = threadIdx.x;
  const float* pkb = pk + (long)b * NN * KD;
  for (int i = tid; i < 128 * KD; i += 128) {
    int m = i >> 6;
    apack[i] = (__bf16)(m < NN ? pkb[i] : 0.0f);
  }
  __syncthreads();

  const int wave = tid >> 5;
  const int lane = tid & 31;
  const int half = lane >> 4;
  const int l15 = lane & 15;
  const long vcol = v0 + wave * 16 + l15;
  const float* mfb = mf + (long)b * CC * VV;

  v8f acc[8] = {};
  for (int kc = 0; kc < 2; ++kc) {
    Frag16 bf;
    const float* mcol = mfb + (long)(kc * 32 + half * 16) * VV + vcol;
#pragma unroll
    for (int j = 0; j < 8; ++j)
      bf.u[j] = pack_bf16x2(mcol[(long)(2 * j) * VV], mcol[(long)(2 * j + 1) * VV]);
#pragma unroll
    for (int mt = 0; mt < 8; ++mt) {
      Frag16 a;
      const __bf16* arow = &apack[(mt * 16 + l15) * KD + kc * 32];
#pragma unroll
      for (int j = 0; j < 8; ++j)
        a.u[j] = *(const unsigned int*)&arow[half * 16 + 2 * j];
      acc[mt] = __builtin_amdgcn_wmma_f32_16x16x32_bf16(
          false, a.v, false, bf.v, (short)0, acc[mt], false, false);
    }
  }

  float s = 0.0f, ss = 0.0f;
  float* pmb = pm + (long)b * NN * VV;
#pragma unroll
  for (int mt = 0; mt < 8; ++mt) {
#pragma unroll
    for (int r = 0; r < 8; ++r) {
      int m = mt * 16 + r + 8 * half;
      float v = acc[mt][r];
      if (m < NN) {
        pmb[(long)m * VV + vcol] = v;
        s += v;
        ss += v * v;
      }
    }
  }

  __shared__ float red[256];
  red[tid] = s;
  red[128 + tid] = ss;
  __syncthreads();
  for (int off = 64; off > 0; off >>= 1) {
    if (tid < off) {
      red[tid] += red[tid + off];
      red[128 + tid] += red[128 + tid + off];
    }
    __syncthreads();
  }
  if (tid == 0) {
    partial[2 * blockIdx.x] = red[0];
    partial[2 * blockIdx.x + 1] = red[128];
  }
}

// =====================================================================
// Kernel 7: finalize global mean / inv-std over all pred_masks (double acc)
// =====================================================================
__global__ __launch_bounds__(256) void finalize_stats_kernel(
    const float* __restrict__ partial, int nparts, float* __restrict__ stats) {
  __shared__ double rs[256], rss[256];
  double s = 0.0, ss = 0.0;
  for (int i = threadIdx.x; i < nparts; i += 256) {
    s += (double)partial[2 * i];
    ss += (double)partial[2 * i + 1];
  }
  rs[threadIdx.x] = s;
  rss[threadIdx.x] = ss;
  __syncthreads();
  for (int off = 128; off > 0; off >>= 1) {
    if (threadIdx.x < off) {
      rs[threadIdx.x] += rs[threadIdx.x + off];
      rss[threadIdx.x] += rss[threadIdx.x + off];
    }
    __syncthreads();
  }
  if (threadIdx.x == 0) {
    double N = (double)BB * NN * VV;
    double mu = rs[0] / N;
    double var = rss[0] / N - mu * mu;
    stats[0] = (float)mu;
    stats[1] = (float)(1.0 / sqrt(var + (double)EPSF));
  }
}

// =====================================================================
// Kernel 8: normalize pred_masks in place (global BN, gamma/beta)
// =====================================================================
__global__ __launch_bounds__(256) void norm_logits_kernel(
    float* __restrict__ pm, const float* __restrict__ stats,
    const float* __restrict__ g, const float* __restrict__ bt, long n) {
  float mu = stats[0], inv = stats[1];
  float gg = g[0], bb = bt[0];
  long stride = (long)gridDim.x * 256;
  for (long i = (long)blockIdx.x * 256 + threadIdx.x; i < n; i += stride)
    pm[i] = (pm[i] - mu) * inv * gg + bb;
}

// =====================================================================
extern "C" void kernel_launch(void* const* d_in, const int* in_sizes, int n_in,
                              void* d_out, int out_size, void* d_ws, size_t ws_size,
                              hipStream_t stream) {
  const float* features = (const float*)d_in[0];
  const float* mask_features = (const float*)d_in[1];
  const float* conv_w = (const float*)d_in[2];
  const float* conv_b = (const float*)d_in[3];
  const float* embed = (const float*)d_in[4];
  const float* mk_w = (const float*)d_in[5];
  const float* mk_b = (const float*)d_in[6];
  const float* nk_gamma = (const float*)d_in[7];
  const float* nk_beta = (const float*)d_in[8];
  const float* nl_gamma = (const float*)d_in[9];
  const float* nl_beta = (const float*)d_in[10];
  const int* init_masks = (const int*)d_in[11];

  // ---- output regions (tuple: x, inst_features, pred_masks, pred_kernel) ----
  float* out_x = (float*)d_out;                      // B*C*V       = 16777216
  float* out_if = out_x + (long)BB * CC * VV;        // B*120*64    = 15360
  float* out_pm = out_if + (long)BB * NN * CC;       // B*120*V     = 31457280
  float* out_pk = out_pm + (long)BB * NN * VV;       // B*120*64    = 15360

  // ---- small workspace layout (bytes) ----
  char* ws = (char*)d_ws;
  unsigned int* wpack = (unsigned int*)(ws + 0);     // 4*9*32*64 u32 = 294912 B
  float* sums = (float*)(ws + 294912);               // B*101*64 f32  = 51712 B
  float* counts = (float*)(ws + 294912 + 51712);     // B*101 f32     = 808 B
  float* pk_raw = (float*)(ws + 348160);             // B*120*64 f32  = 61440 B
  float* partial = (float*)(ws + 409600);            // 4096*2 f32    = 32768 B
  float* stats = (float*)(ws + 442368);              // 2 f32

  // 0) pack weights to bf16 dword-pairs
  pack_weights_kernel<<<(NCONV * 9 * (CC / 2) * CC + 255) / 256, 256, 0, stream>>>(
      conv_w, wpack);

  // 1) conv stack; ping-pong through out_pm region (scratch) and out_x.
  const int convGrid = BB * DD * HH * (WW / 16);     // 16384
  const int WPL = 9 * 32 * CC;                       // dwords per layer
  conv3x3_wmma_kernel<<<convGrid, 128, 0, stream>>>(
      features, wpack + 0 * WPL, conv_b + 0 * CC, out_pm);
  conv3x3_wmma_kernel<<<convGrid, 128, 0, stream>>>(
      out_pm, wpack + 1 * WPL, conv_b + 1 * CC, out_x);
  conv3x3_wmma_kernel<<<convGrid, 128, 0, stream>>>(
      out_x, wpack + 2 * WPL, conv_b + 2 * CC, out_pm);
  conv3x3_wmma_kernel<<<convGrid, 128, 0, stream>>>(
      out_pm, wpack + 3 * WPL, conv_b + 3 * CC, out_x);

  // 2) segment pooling (zero accumulators first; sums+counts contiguous)
  hipMemsetAsync(sums, 0, 51712 + 808, stream);
  seg_pool_kernel<<<BB * (VV / 2048), 256, 0, stream>>>(
      out_x, init_masks, sums, counts);

  // 3) inst_features
  inst_features_kernel<<<(BB * NN * CC + 255) / 256, 256, 0, stream>>>(
      sums, counts, embed, out_if);

  // 4) pred_kernel linear (raw)
  mask_kernel_linear_kernel<<<(BB * NN * KD + 255) / 256, 256, 0, stream>>>(
      out_if, mk_w, mk_b, pk_raw);

  // 5) BatchNorm1d(KD) -> normalized pred_kernel (final output region)
  norm_kernel_bn_kernel<<<1, 64, 0, stream>>>(pk_raw, nk_gamma, nk_beta, out_pk);

  // 6) BMM via WMMA + fused stats partials
  const int bmmGrid = BB * (VV / 64);                // 4096
  bmm_wmma_kernel<<<bmmGrid, 128, 0, stream>>>(
      out_pk, mask_features, out_pm, partial);

  // 7) finalize global BN stats
  finalize_stats_kernel<<<1, 256, 0, stream>>>(partial, bmmGrid, stats);

  // 8) normalize pred_masks in place
  norm_logits_kernel<<<8192, 256, 0, stream>>>(
      out_pm, stats, nl_gamma, nl_beta, (long)BB * NN * VV);
}
